// NGCF_67018669687009
// MI455X (gfx1250) — compile-verified, matrix operand order checked by reading, not run on hardware
//
#include <hip/hip_runtime.h>
#include <hip/hip_bf16.h>

#define N_USER 50000
#define N_ITEM 100000
#define NTOT   150000
#define EMB    128
#define NNZ    2000000
#define BATCH  4096
#define SLOPE  0.2f
#define MTILES 5                       // 16-row tiles per block (80 rows)
#define ROWS_PER_BLOCK (MTILES * 16)

typedef __bf16 bf16x16 __attribute__((ext_vector_type(16)));
typedef float  f32x8   __attribute__((ext_vector_type(8)));

// ---------------------------------------------------------------------------
// Copy user_emb || item_emb -> ego (float4 granularity)
// ---------------------------------------------------------------------------
__global__ void init_ego(const float4* __restrict__ u4, const float4* __restrict__ i4,
                         float4* __restrict__ ego4, int n4u, int n4tot) {
    int i = blockIdx.x * blockDim.x + threadIdx.x;
    if (i >= n4tot) return;
    ego4[i] = (i < n4u) ? u4[i] : i4[i - n4u];
}

// ---------------------------------------------------------------------------
// Pack one 128x128 f32 weight matrix into the WMMA B-fragment bf16 layout.
// Record r = (tile_n*4 + k32)*32 + lane holds the 16 bf16 this lane feeds to
// v_wmma_f32_16x16x32_bf16:  col n = tile_n*16 + lane%16,
// elems 0..7 -> K = k32*32 + kb + 0..7, elems 8..15 -> +16  (kb = 8 if lane>=16)
// ---------------------------------------------------------------------------
__global__ void pack_w(const float* __restrict__ W, __bf16* __restrict__ out) {
    int t = blockIdx.x * blockDim.x + threadIdx.x;      // 0..1023 records
    if (t >= 8 * 4 * 32) return;
    int lane   = t & 31;
    int k32    = (t >> 5) & 3;
    int tile_n = t >> 7;
    int n  = tile_n * 16 + (lane & 15);
    int kb = (lane >= 16) ? 8 : 0;
    __bf16* o = out + t * 16;
#pragma unroll
    for (int i = 0; i < 16; ++i) {
        int klocal = kb + ((i < 8) ? i : (i + 8));
        int k = k32 * 32 + klocal;
        o[i] = (__bf16)W[k * EMB + n];
    }
}

// ---------------------------------------------------------------------------
// SpMM scatter: side[row] += val * ego[col].  One wave32 per edge; each lane
// owns 4 dims -> coalesced float4 gather + 4 native f32 global atomics.
// ---------------------------------------------------------------------------
__global__ void spmm(const int* __restrict__ arow, const int* __restrict__ acol,
                     const float* __restrict__ aval,
                     const float* __restrict__ ego, float* __restrict__ side) {
    int wid  = (blockIdx.x * blockDim.x + threadIdx.x) >> 5;
    int lane = threadIdx.x & 31;
    int nw   = (gridDim.x * blockDim.x) >> 5;
    for (int e = wid; e < NNZ; e += nw) {
        int   r = arow[e];
        int   c = acol[e];
        float v = aval[e];
        if (e + nw < NNZ) {                         // hide L2 latency of next gather
            __builtin_prefetch(ego + (size_t)acol[e + nw] * EMB + lane * 4, 0, 0);
        }
        float4 x = *(const float4*)(ego + (size_t)c * EMB + lane * 4);
        float* d = side + (size_t)r * EMB + lane * 4;
        unsafeAtomicAdd(d + 0, v * x.x);
        unsafeAtomicAdd(d + 1, v * x.y);
        unsafeAtomicAdd(d + 2, v * x.z);
        unsafeAtomicAdd(d + 3, v * x.w);
    }
}

// ---------------------------------------------------------------------------
// Fused dense layer: ego_out = leaky_relu(side@Wg + bg + (ego.*side)@Wb + bb)
// Block = 80 rows x 128 cols; 8 waves; 5 M-tiles of 16 rows.
//   B-fragments (both weight matrices) are loaded ONCE into registers.
//   Per M-tile: waves 0-3 stage the side tile, waves 4-7 the (ego.*side)
//   tile, as fragment-native bf16 in double-buffered LDS (1 barrier/tile),
//   then every wave runs 8x v_wmma_f32_16x16x32_bf16 from LDS A-frags.
// ---------------------------------------------------------------------------
__global__ __launch_bounds__(256)
void dense_layer(const float* __restrict__ side, const float* __restrict__ ego_in,
                 float* __restrict__ ego_out,
                 const __bf16* __restrict__ wg_pack, const __bf16* __restrict__ wb_pack,
                 const float* __restrict__ bg, const float* __restrict__ bb) {
    // double-buffered: [buf][mat][k32][lane][16] = 2 * 8 KB
    __shared__ __align__(32) __bf16 ldsA[2][2 * 4 * 32 * 16];

    const int t      = threadIdx.x;
    const int lane   = t & 31;
    const int tile_n = t >> 5;                 // 0..7 (wave id)
    const int row0   = blockIdx.x * ROWS_PER_BLOCK;

    // ---- B-fragments: load once, keep in VGPRs for all 5 M-tiles ----
    const bf16x16* gp = (const bf16x16*)wg_pack;
    const bf16x16* bp = (const bf16x16*)wb_pack;
    bf16x16 FG[4], FB[4];
#pragma unroll
    for (int k32 = 0; k32 < 4; ++k32) {
        FG[k32] = gp[(tile_n * 4 + k32) * 32 + lane];
        FB[k32] = bp[(tile_n * 4 + k32) * 32 + lane];
    }

    // staging coordinates (wave-uniform mat split: waves 0-3 side, 4-7 ego.*side)
    const int mat   = t >> 7;
    const int k32s  = (t >> 5) & 3;
    const int srow  = lane & 15;               // staged A row within tile
    const int kb    = (lane >= 16) ? 8 : 0;
    const int base  = k32s * 32 + kb;
    const int ldrec = ((mat * 4 + k32s) * 32 + lane) * 16;

    // epilogue coordinates
    const int colg = tile_n * 16 + (lane & 15);
    const float bgv = bg[colg];
    const float bbv = bb[colg];
    const int rsub = (lane >= 16) ? 8 : 0;

    const bf16x16* lA0 = (const bf16x16*)&ldsA[0][0];
    const bf16x16* lA1 = (const bf16x16*)&ldsA[1][0];

    for (int mt = 0; mt < MTILES; ++mt) {
        const int r0 = row0 + mt * 16;

        // ---- stage A fragments for this tile into buffer mt&1 ----
        {
            const float* sp = side + (size_t)(r0 + srow) * EMB + base;
            float4 sv[4];
            sv[0] = *(const float4*)(sp);
            sv[1] = *(const float4*)(sp + 4);
            sv[2] = *(const float4*)(sp + 16);
            sv[3] = *(const float4*)(sp + 20);
            if (mat) {
                const float* ep = ego_in + (size_t)(r0 + srow) * EMB + base;
                float4 ev[4];
                ev[0] = *(const float4*)(ep);
                ev[1] = *(const float4*)(ep + 4);
                ev[2] = *(const float4*)(ep + 16);
                ev[3] = *(const float4*)(ep + 20);
                float* sf = (float*)sv;
                const float* ef = (const float*)ev;
#pragma unroll
                for (int i = 0; i < 16; ++i) sf[i] *= ef[i];
            }
            const float* sf = (const float*)sv;
            bf16x16 a;
#pragma unroll
            for (int i = 0; i < 16; ++i) a[i] = (__bf16)sf[i];
            *(bf16x16*)&ldsA[mt & 1][ldrec] = a;
        }
        __syncthreads();

        // ---- 8 WMMAs: A from LDS, B from registers ----
        const bf16x16* lA = (mt & 1) ? lA1 : lA0;
        f32x8 accG = {};
        f32x8 accB = {};
#pragma unroll
        for (int k32 = 0; k32 < 4; ++k32) {
            bf16x16 aG = lA[(0 * 4 + k32) * 32 + lane];
            bf16x16 aB = lA[(1 * 4 + k32) * 32 + lane];
            accG = __builtin_amdgcn_wmma_f32_16x16x32_bf16(false, aG, false, FG[k32],
                                                           (short)0, accG, false, false);
            accB = __builtin_amdgcn_wmma_f32_16x16x32_bf16(false, aB, false, FB[k32],
                                                           (short)0, accB, false, false);
        }

        // ---- epilogue: bias + leaky_relu, scatter 16x16 tile ----
        const int rbase = r0 + rsub;
#pragma unroll
        for (int i = 0; i < 8; ++i) {
            float v = accG[i] + accB[i] + bgv + bbv;
            v = (v > 0.f) ? v : SLOPE * v;
            ego_out[(size_t)(rbase + i) * EMB + colg] = v;
        }
    }
}

// ---------------------------------------------------------------------------
// Gather one 128-col slice of the output: rows = users | N_USER+pos | N_USER+neg,
// optional on-the-fly L2 normalization (wave shuffle reduction over 128 dims).
// ---------------------------------------------------------------------------
__global__ void gather_rows(const float* __restrict__ src,
                            const int* __restrict__ users, const int* __restrict__ pos,
                            const int* __restrict__ neg,
                            float* __restrict__ out, int col_off, int do_norm) {
    int gw   = (blockIdx.x * blockDim.x + threadIdx.x) >> 5;
    int lane = threadIdx.x & 31;
    if (gw >= 3 * BATCH) return;
    int seg = gw / BATCH;
    int j   = gw - seg * BATCH;
    int node = (seg == 0) ? users[j] : ((seg == 1) ? N_USER + pos[j] : N_USER + neg[j]);
    float4 v = *(const float4*)(src + (size_t)node * EMB + lane * 4);
    float scale = 1.f;
    if (do_norm) {
        float ss = v.x * v.x + v.y * v.y + v.z * v.z + v.w * v.w;
#pragma unroll
        for (int m = 16; m; m >>= 1) ss += __shfl_xor(ss, m, 32);
        scale = 1.f / fmaxf(sqrtf(ss), 1e-12f);
    }
    float4 o = { v.x * scale, v.y * scale, v.z * scale, v.w * scale };
    *(float4*)(out + (size_t)gw * 512 + col_off + lane * 4) = o;
}

// ---------------------------------------------------------------------------
extern "C" void kernel_launch(void* const* d_in, const int* in_sizes, int n_in,
                              void* d_out, int out_size, void* d_ws, size_t ws_size,
                              hipStream_t stream) {
    const int*   users = (const int*)d_in[0];
    const int*   pos   = (const int*)d_in[1];
    const int*   neg   = (const int*)d_in[2];
    const int*   arow  = (const int*)d_in[3];
    const int*   acol  = (const int*)d_in[4];
    const float* aval  = (const float*)d_in[5];
    const float* uemb  = (const float*)d_in[6];
    const float* iemb  = (const float*)d_in[7];
    float* out = (float*)d_out;

    char* ws = (char*)d_ws;
    const size_t EGO_BYTES = (size_t)NTOT * EMB * sizeof(float);   // 76.8 MB
    float*  ego0  = (float*)(ws);
    float*  ego1  = (float*)(ws + EGO_BYTES);
    float*  side  = (float*)(ws + 2 * EGO_BYTES);
    __bf16* wpack = (__bf16*)(ws + 3 * EGO_BYTES);                 // 6 x 16384 bf16

    // Pack all 6 weight matrices into WMMA B-fragment layout (bf16).
    for (int l = 0; l < 3; ++l) {
        pack_w<<<4, 256, 0, stream>>>((const float*)d_in[8 + 4 * l],  wpack + (2 * l) * 16384);
        pack_w<<<4, 256, 0, stream>>>((const float*)d_in[10 + 4 * l], wpack + (2 * l + 1) * 16384);
    }

    // ego0 = concat(user_emb, item_emb)
    {
        int n4u = N_USER * (EMB / 4);
        int n4t = NTOT * (EMB / 4);
        init_ego<<<(n4t + 255) / 256, 256, 0, stream>>>(
            (const float4*)uemb, (const float4*)iemb, (float4*)ego0, n4u, n4t);
    }

    // Layer-0 slice of the output (raw, unnormalized embeddings).
    gather_rows<<<(3 * BATCH * 32 + 255) / 256, 256, 0, stream>>>(
        ego0, users, pos, neg, out, 0, 0);

    float* cur = ego0;
    float* nxt = ego1;
    for (int l = 0; l < 3; ++l) {
        hipMemsetAsync(side, 0, EGO_BYTES, stream);
        spmm<<<2048, 256, 0, stream>>>(arow, acol, aval, cur, side);
        dense_layer<<<NTOT / ROWS_PER_BLOCK, 256, 0, stream>>>(
            side, cur, nxt,
            wpack + (2 * l) * 16384, wpack + (2 * l + 1) * 16384,
            (const float*)d_in[9 + 4 * l], (const float*)d_in[11 + 4 * l]);
        gather_rows<<<(3 * BATCH * 32 + 255) / 256, 256, 0, stream>>>(
            nxt, users, pos, neg, out, (l + 1) * 128, 1);
        float* t = cur; cur = nxt; nxt = t;
    }
}
